// StyleConvBase3D_47742856462501
// MI455X (gfx1250) — matrix operand order repaired and checked by reading, not compiled, not run on hardware
//
#include <hip/hip_runtime.h>

// ---------------------------------------------------------------------------
// StyleConv3D (modulated conv) as implicit GEMM on CDNA5 WMMA.
//   Per batch b:  Y[b] (Cout=32 x N=46^3) = Wn[b] (32 x 864) x im2col(x[b])
//   FP16 operands, FP32 accumulation via v_wmma_f32_16x16x32_f16.
// Round 2: 2 z-rows per WG -> 4-slice slab (1.5x less staging/row), each
// wave computes two 16x16 tiles sharing one A-fragment stream (54 WMMAs).
// ---------------------------------------------------------------------------

typedef __attribute__((ext_vector_type(16))) _Float16 v16h;
typedef __attribute__((ext_vector_type(8)))  float    v8f;

#define CIN   32
#define COUT  32
#define DIN   48
#define DOUT  46
#define ZBLK  2                  // output z-rows per workgroup (46 % 2 == 0)
#define ZSL   (ZBLK + 2)         // input z-slices staged = 4
#define KT    27                 // 3x3x3 taps
#define KSTEPS 27                // K = 864 = 27 chunks of 32 (one tap, all ci)
#define AP_PER_MTILE (KSTEPS*32*16)   // 13824 f16
#define AP_PER_B     (2*AP_PER_MTILE) // 27648 f16 per batch
#define XROW  40                 // padded ci-row (f16): 80B stride, 16B aligned

union Frag {                     // 32B fragment: two ds/global b128 loads
    uint4 q[2];
    v16h  h;
};

// ---------------------------------------------------------------------------
// Kernel 1: style modulation + demodulation, pack weights into the exact
// per-lane A-fragment order used here for V_WMMA_F32_16X16X32_F16 (K contiguous per lane half, not the ISA table).
// K ordering: k = tap*32 + ci  (chunk t == tap, ci across the 32 K-slots).
// AP[b][mtile][t][lane][j] : lane's 16 contiguous f16 = its A fragment.
// ---------------------------------------------------------------------------
__global__ __launch_bounds__(256)
void style_pack_weights(const float* __restrict__ s,
                        const float* __restrict__ style_w,
                        const float* __restrict__ style_b,
                        const float* __restrict__ weight,
                        _Float16* __restrict__ AP)
{
    const int b   = blockIdx.x;
    const int tid = threadIdx.x;
    __shared__ float smod[CIN];
    __shared__ float inv[COUT];
    __shared__ float part[COUT][8];

    if (tid < CIN) {
        smod[tid] = style_b[tid]
                  + s[b*2 + 0] * style_w[tid*2 + 0]
                  + s[b*2 + 1] * style_w[tid*2 + 1];
    }
    __syncthreads();

    {   // demod norm: sum over (ci, tap) of (w * smod)^2, 8 partials per co
        const int co = tid >> 3, p = tid & 7;
        float acc = 0.f;
        for (int kk = p; kk < CIN*KT; kk += 8) {
            const int ci = kk / KT, t = kk % KT;
            const float w = weight[(co*CIN + ci)*KT + t] * smod[ci];
            acc += w * w;
        }
        part[co][p] = acc;
    }
    __syncthreads();
    if (tid < COUT) {
        float sum = 0.f;
#pragma unroll
        for (int p = 0; p < 8; ++p) sum += part[tid][p];
        inv[tid] = 1.0f / sqrtf(sum + 1e-8f);
    }
    __syncthreads();

    _Float16* apb = AP + b*AP_PER_B;
    for (int e = tid; e < AP_PER_B; e += blockDim.x) {
        const int mtile = e / AP_PER_MTILE;
        const int r     = e % AP_PER_MTILE;
        const int t     = r >> 9;          // / 512
        const int r2    = r & 511;
        const int lane  = r2 >> 4;
        const int j     = r2 & 15;
        // A-layout (16-bit 16x32): lane half selects contiguous K block of 16,
        // element j walks K by 1 (matches B fragment ci order).
        const int hi = lane >> 4, m = lane & 15;
        const int co = mtile*16 + m;
        const int ci = hi*16 + j;
        const float w = weight[(co*CIN + ci)*KT + t] * smod[ci] * inv[co];
        apb[e] = (_Float16)w;
    }
}

// ---------------------------------------------------------------------------
// Kernel 2: one WG per (b, zo-pair, yo). 6 waves = 2 Cout-tiles x 3 x-tiles;
// each wave produces two 16x16 tiles (z rows zo, zo+1) from one A stream.
// LDS slab: x[4z][3y][50x][32ci] f16, ci innermost -> B fragments are two
// contiguous ds_load_b128 per lane (lane half = K block of 16 = ci block).
// ---------------------------------------------------------------------------
__global__ __launch_bounds__(192)
void style_conv3d_wmma(const float* __restrict__ x,
                       const _Float16* __restrict__ AP,
                       const float* __restrict__ bias,
                       float* __restrict__ out)
{
    // ZSL*3 rows of 50 x-slots (2 pad slots so n+kx<=49 reads stay in-bounds)
    __shared__ alignas(16) _Float16 xs[ZSL*3*50*XROW];   // 48000 B

    const int tid = threadIdx.x;
    const int bid = blockIdx.x;
    const int b   = bid / ((DOUT/ZBLK)*DOUT);
    const int rr  = bid % ((DOUT/ZBLK)*DOUT);
    const int zo  = (rr / DOUT) * ZBLK;
    const int yo  = rr % DOUT;

    // ---- stage slab: 32ci x 4z x 3y x 48x, f32 -> f16, transpose ci inner
    const float* xb = x + (b*CIN)*(DIN*DIN*DIN);
#pragma unroll 4
    for (int it = 0; it < 96; ++it) {                  // ZSL*3*32*48/192 = 96
        const int e  = tid + it*192;
        const int xx = e % DIN;                        // innermost: coalesced
        const int t2 = e / DIN;
        const int ci = t2 & 31;
        const int zy = t2 >> 5;                        // 0..11 = zz*3+yy
        const int zz = zy / 3, yy = zy % 3;
        const float v = xb[((ci*DIN + (zo + zz))*DIN + (yo + yy))*DIN + xx];
        xs[(zy*50 + xx)*XROW + ci] = (_Float16)v;
    }
    __syncthreads();

    const int lane  = tid & 31;
    const int wv    = tid >> 5;        // 0..5
    const int mtile = wv & 1;          // Cout half
    const int ntile = wv >> 1;         // 0..2: 16 x-positions each
    const int n     = lane & 15;       // N column
    const int khalf = lane >> 4;       // K(=ci) block of 16 (contiguous order, matches A pack)

    const _Float16* aBase = AP + (b*2 + mtile)*AP_PER_MTILE + lane*16;
    __builtin_prefetch(aBase, 0, 3);   // global_prefetch_b8: warm L2/WGP$

    v8f acc0 = {};
    v8f acc1 = {};
#pragma unroll
    for (int t = 0; t < KSTEPS; ++t) {
        const int kz = t / 9, ky = (t / 3) % 3, kx = t % 3;
        Frag a, b0, b1;
        const uint4* ap = (const uint4*)(aBase + t*512);   // L2-hot weights
        a.q[0] = ap[0];
        a.q[1] = ap[1];
        const int col = ntile*16 + n + kx;
        const _Float16* bp0 =
            &xs[(((kz+0)*3 + ky)*50 + col)*XROW + khalf*16];
        const _Float16* bp1 =
            &xs[(((kz+1)*3 + ky)*50 + col)*XROW + khalf*16];
        b0.q[0] = ((const uint4*)bp0)[0];                  // ds_load_b128 x2
        b0.q[1] = ((const uint4*)bp0)[1];
        b1.q[0] = ((const uint4*)bp1)[0];
        b1.q[1] = ((const uint4*)bp1)[1];
        acc0 = __builtin_amdgcn_wmma_f32_16x16x32_f16(
                   false, a.h, false, b0.h, (short)0, acc0, false, false);
        acc1 = __builtin_amdgcn_wmma_f32_16x16x32_f16(
                   false, a.h, false, b1.h, (short)0, acc1, false, false);
    }

    // C/D layout: acc[v] -> M = khalf*8 + v, lane n -> N column
    const int xo = ntile*16 + n;
    if (xo < DOUT) {                   // mask padded columns (garbage-safe)
#pragma unroll
        for (int v = 0; v < 8; ++v) {
            const int co = mtile*16 + khalf*8 + v;
            const float bv = bias[co];
            const int base = ((b*COUT + co)*DOUT + zo)*DOUT*DOUT + yo*DOUT + xo;
            out[base]             = acc0[v] + bv;
            out[base + DOUT*DOUT] = acc1[v] + bv;   // z row zo+1
        }
    }
}

// ---------------------------------------------------------------------------
extern "C" void kernel_launch(void* const* d_in, const int* in_sizes, int n_in,
                              void* d_out, int out_size, void* d_ws, size_t ws_size,
                              hipStream_t stream)
{
    const float* x   = (const float*)d_in[0];   // (8,32,48,48,48)
    const float* s   = (const float*)d_in[1];   // (8,2)
    const float* sw  = (const float*)d_in[2];   // (32,2)
    const float* sb  = (const float*)d_in[3];   // (32,)
    const float* w   = (const float*)d_in[4];   // (32,32,3,3,3)
    const float* bi  = (const float*)d_in[5];   // (32,)
    float* out = (float*)d_out;                 // (8,32,46,46,46) f32
    _Float16* AP = (_Float16*)d_ws;             // 8*27648 f16 = 442 KB

    style_pack_weights<<<8, 256, 0, stream>>>(s, sw, sb, w, AP);
    style_conv3d_wmma<<<8*(DOUT/ZBLK)*DOUT, 192, 0, stream>>>(x, AP, bi, out);
}
